// CausalSelfAttention_18580028522616
// MI455X (gfx1250) — matrix operand order repaired and checked by reading, not compiled, not run on hardware
//
#include <hip/hip_runtime.h>
#include <stdint.h>

typedef _Float16 v16h __attribute__((ext_vector_type(16)));
typedef float    v8f  __attribute__((ext_vector_type(8)));

union Frag16 { v16h v; _Float16 h[16]; uint32_t u[8]; };
union FragC  { v8f  v; float f[8]; };

#define WMMA16(A, B, Cacc) \
  __builtin_amdgcn_wmma_f32_16x16x32_f16(false, (A), false, (B), (short)0, (Cacc), false, false)

// Problem constants (match reference)
#define BB   2
#define TT   2048
#define CC   1024
#define HH   16
#define HSZ  64
#define N3   3072

// ---------------------------------------------------------------------------
// fp32 -> f16 elementwise
__global__ void k_cvt_f16(const float* __restrict__ in, _Float16* __restrict__ out, int n) {
  int i = blockIdx.x * blockDim.x + threadIdx.x;
  if (i < n) out[i] = (_Float16)in[i];
}

// W[K][N] fp32 -> Wt[N][K] f16 (coalesced writes)
__global__ void k_transcvt(const float* __restrict__ W, _Float16* __restrict__ Wt, int K, int N) {
  int i = blockIdx.x * blockDim.x + threadIdx.x;
  if (i >= K * N) return;
  int n = i / K, k = i - n * K;
  Wt[i] = (_Float16)W[(long)k * N + n];
}

// ---------------------------------------------------------------------------
// QKV GEMM:  [4096 x 1024] @ [1024 x 3072] (+bias), fused split/scale/layout.
// Block: 256 thr = 8 waves (2 M x 4 N), wave tile 32x64 -> block tile 64x256.
__global__ __launch_bounds__(256) void k_qkv(const _Float16* __restrict__ Xh,
                                             const _Float16* __restrict__ WaT,
                                             const float* __restrict__ bias,
                                             _Float16* __restrict__ Qs,
                                             _Float16* __restrict__ Ks,
                                             _Float16* __restrict__ Vt) {
  const int lane = threadIdx.x & 31;
  const int wid  = threadIdx.x >> 5;
  const int wm   = wid & 1, wn = wid >> 1;
  const int m0   = blockIdx.y * 64  + wm * 32;
  const int n0   = blockIdx.x * 256 + wn * 64;
  const int c15  = lane & 15, grp = lane >> 4;

  FragC acc[2][4];
#pragma unroll
  for (int mi = 0; mi < 2; ++mi)
#pragma unroll
    for (int ni = 0; ni < 4; ++ni) acc[mi][ni].v = (v8f){};

  for (int kb = 0; kb < CC; kb += 32) {
    Frag16 a[2];
#pragma unroll
    for (int mi = 0; mi < 2; ++mi) {
      const _Float16* row = Xh + (long)(m0 + mi * 16 + c15) * CC + kb;
#pragma unroll
      for (int v = 0; v < 8; ++v) {
        int k = 2 * v + 8 * ((v >> 2) + grp);
        a[mi].u[v] = *(const uint32_t*)(row + k);
      }
    }
    Frag16 bf[4];
#pragma unroll
    for (int ni = 0; ni < 4; ++ni) {
      const _Float16* row = WaT + (long)(n0 + ni * 16 + c15) * CC + kb;
#pragma unroll
      for (int v = 0; v < 8; ++v) {
        int k = 2 * v + 8 * ((v >> 2) + grp);
        bf[ni].u[v] = *(const uint32_t*)(row + k);
      }
    }
#pragma unroll
    for (int mi = 0; mi < 2; ++mi)
#pragma unroll
      for (int ni = 0; ni < 4; ++ni)
        acc[mi][ni].v = WMMA16(a[mi].v, bf[ni].v, acc[mi][ni].v);
  }

#pragma unroll
  for (int mi = 0; mi < 2; ++mi)
#pragma unroll
    for (int ni = 0; ni < 4; ++ni)
#pragma unroll
      for (int r = 0; r < 8; ++r) {
        int m = m0 + mi * 16 + r + 8 * grp;        // global row = b*T + t
        int n = n0 + ni * 16 + c15;                // global col in [0,3C)
        float val = acc[mi][ni].f[r] + bias[n];
        int sec = n >> 10, c0 = n & 1023;
        int h = c0 >> 6, d = c0 & 63;
        int b = m >> 11, t = m & 2047;
        long bh = (long)(b * HH + h);
        if (sec == 0)      Qs[(bh * TT + t) * HSZ + d] = (_Float16)(val * 0.125f); // fold 1/sqrt(64)
        else if (sec == 1) Ks[(bh * TT + t) * HSZ + d] = (_Float16)val;
        else               Vt[(bh * HSZ + d) * TT + t] = (_Float16)val;            // transposed V
      }
}

// ---------------------------------------------------------------------------
// Flash-style causal attention. 1 wave = 16 query rows; 4 waves / block.
__global__ __launch_bounds__(128) void k_attn(const _Float16* __restrict__ Qs,
                                              const _Float16* __restrict__ Ks,
                                              const _Float16* __restrict__ Vt,
                                              _Float16* __restrict__ Yh) {
  __shared__ float Pf[4][16 * 32];   // per-wave P staging (C-layout -> A-layout)
  const int lane = threadIdx.x & 31;
  const int wid  = threadIdx.x >> 5;
  const int g    = blockIdx.x * 4 + wid;      // q-tile id, 4096 total
  const int bh   = g >> 7;                    // b*H + h
  const int qt   = g & 127;
  const int b    = bh >> 4, h = bh & 15;
  const int qbase = qt * 16;
  const int c15  = lane & 15, grp = lane >> 4;

  const _Float16* Qb = Qs + (long)bh * TT * HSZ;
  const _Float16* Kb = Ks + (long)bh * TT * HSZ;
  const _Float16* Vb = Vt + (long)bh * HSZ * TT;

  Frag16 aq[2];
#pragma unroll
  for (int s = 0; s < 2; ++s) {
    const _Float16* row = Qb + (long)(qbase + c15) * HSZ + 32 * s;
#pragma unroll
    for (int v = 0; v < 8; ++v) {
      int k = 2 * v + 8 * ((v >> 2) + grp);
      aq[s].u[v] = *(const uint32_t*)(row + k);
    }
  }

  FragC O[4];
#pragma unroll
  for (int nt = 0; nt < 4; ++nt) O[nt].v = (v8f){};
  float mrow[8], lrow[8];
#pragma unroll
  for (int r = 0; r < 8; ++r) { mrow[r] = -1e30f; lrow[r] = 0.f; }

  const int nj = (qbase + 47) >> 5;           // 32-key blocks needed (causal)
  for (int j = 0; j < nj; ++j) {
    const int kb = j * 32;
    // K^T fragments: two 16-key tiles x two 32-deep k-steps
    Frag16 bk[2][2];
#pragma unroll
    for (int t2 = 0; t2 < 2; ++t2) {
      const _Float16* row = Kb + (long)(kb + t2 * 16 + c15) * HSZ;
#pragma unroll
      for (int s = 0; s < 2; ++s)
#pragma unroll
        for (int v = 0; v < 8; ++v) {
          int kd = 2 * v + 8 * ((v >> 2) + grp) + 32 * s;
          bk[t2][s].u[v] = *(const uint32_t*)(row + kd);
        }
    }
    FragC S0, S1;
    S0.v = WMMA16(aq[0].v, bk[0][0].v, (v8f){});
    S0.v = WMMA16(aq[1].v, bk[0][1].v, S0.v);
    S1.v = WMMA16(aq[0].v, bk[1][0].v, (v8f){});
    S1.v = WMMA16(aq[1].v, bk[1][1].v, S1.v);

    // causal mask + online softmax (rows live across 16 lanes at fixed vgpr)
    float nmax[8];
#pragma unroll
    for (int r = 0; r < 8; ++r) {
      int qg = qbase + r + 8 * grp;
      float s0 = (kb + c15      > qg) ? -1e30f : S0.f[r];
      float s1 = (kb + 16 + c15 > qg) ? -1e30f : S1.f[r];
      S0.f[r] = s0; S1.f[r] = s1;
      float mx = fmaxf(s0, s1);
#pragma unroll
      for (int off = 1; off < 16; off <<= 1) mx = fmaxf(mx, __shfl_xor(mx, off, 32));
      nmax[r] = fmaxf(mrow[r], mx);
    }
#pragma unroll
    for (int r = 0; r < 8; ++r) {
      float sc = __expf(mrow[r] - nmax[r]);
      mrow[r] = nmax[r];
      float p0 = __expf(S0.f[r] - nmax[r]);
      float p1 = __expf(S1.f[r] - nmax[r]);
      int row = r + 8 * grp;
      Pf[wid][row * 32 + c15]      = p0;   // same-wave DS ops are in-order: no barrier
      Pf[wid][row * 32 + 16 + c15] = p1;
      float sm = p0 + p1;
#pragma unroll
      for (int off = 1; off < 16; off <<= 1) sm += __shfl_xor(sm, off, 32);
      lrow[r] = lrow[r] * sc + sm;
#pragma unroll
      for (int nt = 0; nt < 4; ++nt) O[nt].f[r] *= sc;
    }
    // reload P in A-fragment layout
    Frag16 pa;
#pragma unroll
    for (int e = 0; e < 16; ++e) {
      int k = e + 8 * ((e >> 3) + grp);
      pa.h[e] = (_Float16)Pf[wid][c15 * 32 + k];
    }
    // P @ V  (V transposed -> contiguous dword B-frag loads)
#pragma unroll
    for (int nt = 0; nt < 4; ++nt) {
      const _Float16* row = Vb + (long)(nt * 16 + c15) * TT + kb;
      Frag16 bv;
#pragma unroll
      for (int v = 0; v < 8; ++v) {
        int kd = 2 * v + 8 * ((v >> 2) + grp);
        bv.u[v] = *(const uint32_t*)(row + kd);
      }
      O[nt].v = WMMA16(pa.v, bv.v, O[nt].v);
    }
  }

#pragma unroll
  for (int nt = 0; nt < 4; ++nt)
#pragma unroll
    for (int r = 0; r < 8; ++r) {
      int t = qbase + r + 8 * grp;
      int d = nt * 16 + c15;
      float o = O[nt].f[r] / lrow[r];
      Yh[((long)(b * TT + t)) * CC + h * HSZ + d] = (_Float16)o;
    }
}

// ---------------------------------------------------------------------------
// Output projection: [4096 x 1024] @ [1024 x 1024] + bias -> fp32 out
__global__ __launch_bounds__(256) void k_proj(const _Float16* __restrict__ Yh,
                                              const _Float16* __restrict__ WpT,
                                              const float* __restrict__ bias,
                                              float* __restrict__ out) {
  const int lane = threadIdx.x & 31;
  const int wid  = threadIdx.x >> 5;
  const int wm   = wid & 1, wn = wid >> 1;
  const int m0   = blockIdx.y * 64  + wm * 32;
  const int n0   = blockIdx.x * 256 + wn * 64;
  const int c15  = lane & 15, grp = lane >> 4;

  FragC acc[2][4];
#pragma unroll
  for (int mi = 0; mi < 2; ++mi)
#pragma unroll
    for (int ni = 0; ni < 4; ++ni) acc[mi][ni].v = (v8f){};

  for (int kb = 0; kb < CC; kb += 32) {
    Frag16 a[2];
#pragma unroll
    for (int mi = 0; mi < 2; ++mi) {
      const _Float16* row = Yh + (long)(m0 + mi * 16 + c15) * CC + kb;
#pragma unroll
      for (int v = 0; v < 8; ++v) {
        int k = 2 * v + 8 * ((v >> 2) + grp);
        a[mi].u[v] = *(const uint32_t*)(row + k);
      }
    }
    Frag16 bf[4];
#pragma unroll
    for (int ni = 0; ni < 4; ++ni) {
      const _Float16* row = WpT + (long)(n0 + ni * 16 + c15) * CC + kb;
#pragma unroll
      for (int v = 0; v < 8; ++v) {
        int k = 2 * v + 8 * ((v >> 2) + grp);
        bf[ni].u[v] = *(const uint32_t*)(row + k);
      }
    }
#pragma unroll
    for (int mi = 0; mi < 2; ++mi)
#pragma unroll
      for (int ni = 0; ni < 4; ++ni)
        acc[mi][ni].v = WMMA16(a[mi].v, bf[ni].v, acc[mi][ni].v);
  }

#pragma unroll
  for (int mi = 0; mi < 2; ++mi)
#pragma unroll
    for (int ni = 0; ni < 4; ++ni)
#pragma unroll
      for (int r = 0; r < 8; ++r) {
        int m = m0 + mi * 16 + r + 8 * grp;
        int n = n0 + ni * 16 + c15;
        out[(long)m * CC + n] = acc[mi][ni].f[r] + bias[n];
      }
}

// ---------------------------------------------------------------------------
extern "C" void kernel_launch(void* const* d_in, const int* in_sizes, int n_in,
                              void* d_out, int out_size, void* d_ws, size_t ws_size,
                              hipStream_t stream) {
  const float* x      = (const float*)d_in[0];
  const float* W_attn = (const float*)d_in[1];
  const float* b_attn = (const float*)d_in[2];
  const float* W_proj = (const float*)d_in[3];
  const float* b_proj = (const float*)d_in[4];
  float* out = (float*)d_out;

  char* ws = (char*)d_ws;
  const long MB = 1024 * 1024;
  _Float16* Xh  = (_Float16*)(ws + 0 * MB);   // 8 MB  [B*T][C]
  _Float16* WaT = (_Float16*)(ws + 8 * MB);   // 6 MB  [3C][C]
  _Float16* WpT = (_Float16*)(ws + 14 * MB);  // 2 MB  [C][C]
  _Float16* Qs  = (_Float16*)(ws + 16 * MB);  // 8 MB  [B,H,T,hs] (scaled)
  _Float16* Ks  = (_Float16*)(ws + 24 * MB);  // 8 MB  [B,H,T,hs]
  _Float16* Vt  = (_Float16*)(ws + 32 * MB);  // 8 MB  [B,H,hs,T]
  _Float16* Yh  = (_Float16*)(ws + 40 * MB);  // 8 MB  [B*T][C]

  const int nx = BB * TT * CC;
  k_cvt_f16<<<(nx + 255) / 256, 256, 0, stream>>>(x, Xh, nx);
  k_transcvt<<<(CC * N3 + 255) / 256, 256, 0, stream>>>(W_attn, WaT, CC, N3);
  k_transcvt<<<(CC * CC + 255) / 256, 256, 0, stream>>>(W_proj, WpT, CC, CC);

  dim3 gq(N3 / 256, (BB * TT) / 64);
  k_qkv<<<gq, 256, 0, stream>>>(Xh, WaT, b_attn, Qs, Ks, Vt);

  k_attn<<<(BB * HH * (TT / 16)) / 4, 128, 0, stream>>>(Qs, Ks, Vt, Yh);

  dim3 gp(CC / 256, (BB * TT) / 64);
  k_proj<<<gp, 256, 0, stream>>>(Yh, WpT, b_proj, out);
}
